// LPD_54125177864521
// MI455X (gfx1250) — matrix operand order repaired and checked by reading, not compiled, not run on hardware
//
#include <hip/hip_runtime.h>
#include <math.h>

typedef __attribute__((ext_vector_type(16))) _Float16 v16h;
typedef __attribute__((ext_vector_type(8)))  _Float16 v8h;
typedef __attribute__((ext_vector_type(8)))  float    v8f;

#define BATCH 4
#define NPTS  4096
#define KNN   20
#define EMB_ELEMS ((size_t)BATCH * 1024 * NPTS)   // 16777216 per net

__device__ __forceinline__ float lrelu(float v) { return v >= 0.f ? v : 0.2f * v; }

// ---- WMMA A-operand pack: 16-bit A 16x32 layout ----------------------------
// lane: M = lane&15, half = lane>>4.  K(e) = base + e, base = kt*32+half*16
__device__ __forceinline__ v16h load_wmma_a(const _Float16* Wrow, int base) {
  v8h lo = *(const v8h*)(Wrow + base);
  v8h hi = *(const v8h*)(Wrow + base + 8);
  v16h a;
#pragma unroll
  for (int i = 0; i < 8; ++i) { a[i] = lo[i]; a[8 + i] = hi[i]; }
  return a;
}

// ======================= weight f32 -> f16 conversion =======================
__global__ void __launch_bounds__(256) prep_f16_kernel(const float* s, _Float16* d, int count) {
  int i = blockIdx.x * 256 + threadIdx.x;
  if (i < count) d[i] = (_Float16)s[i];
}

// ======================= conv1 (3->64) + conv2 (64->64) =====================
// writes h2t point-major f16 [b][n][64] and xx[b*N+n] = sum(h2^2) in f32
__global__ void __launch_bounds__(256) conv12_kernel(
    const float* __restrict__ x, const float* __restrict__ w1, const float* __restrict__ b1,
    const float* __restrict__ w2, const float* __restrict__ b2,
    _Float16* __restrict__ h2t, float* __restrict__ xx) {
  __shared__ float W2s[64 * 64];
  __shared__ float W1s[64 * 3];
  __shared__ float B1s[64], B2s[64];
  int tid = threadIdx.x;
  for (int i = tid; i < 64 * 64; i += 256) W2s[i] = w2[i];
  for (int i = tid; i < 64 * 3;  i += 256) W1s[i] = w1[i];
  for (int i = tid; i < 64;      i += 256) { B1s[i] = b1[i]; B2s[i] = b2[i]; }
  __syncthreads();

  int gid = blockIdx.x * 256 + tid;          // 0 .. B*N-1
  int b = gid >> 12, n = gid & (NPTS - 1);
  const float* X = x + (size_t)b * 3 * NPTS;
  float x0 = X[n], x1 = X[NPTS + n], x2 = X[2 * NPTS + n];

  float h1[64];
#pragma unroll 4
  for (int o = 0; o < 64; ++o)
    h1[o] = lrelu(W1s[o * 3] * x0 + W1s[o * 3 + 1] * x1 + W1s[o * 3 + 2] * x2 + B1s[o]);

  float ss = 0.f;
  _Float16* H = h2t + (size_t)gid * 64;
  for (int o = 0; o < 64; ++o) {
    float v = B2s[o];
#pragma unroll 8
    for (int c = 0; c < 64; ++c) v += W2s[o * 64 + c] * h1[c];
    v = lrelu(v);
    ss += v * v;
    H[o] = (_Float16)v;
  }
  xx[gid] = ss;
}

// ======================= top-K smallest selection helper ====================
__device__ void select_min_k(float* dist, float* rd, int* ri, int tid, int* out, int kk, int n) {
  for (int t = 0; t < kk; ++t) {
    float bd = 3.0e38f; int bi = 0x7fffffff;
    for (int j = tid; j < n; j += 256) {
      float d = dist[j];
      if (d < bd || (d == bd && j < bi)) { bd = d; bi = j; }
    }
    rd[tid] = bd; ri[tid] = bi;
    __syncthreads();
    for (int s = 128; s > 0; s >>= 1) {
      if (tid < s) {
        float d2 = rd[tid + s]; int i2 = ri[tid + s];
        if (d2 < rd[tid] || (d2 == rd[tid] && i2 < ri[tid])) { rd[tid] = d2; ri[tid] = i2; }
      }
      __syncthreads();
    }
    if (tid == 0) { out[t] = ri[0]; dist[ri[0]] = 3.0e38f; }
    __syncthreads();
  }
}

// ======================= kNN over 64-dim features ===========================
__global__ void __launch_bounds__(256) knn64_kernel(
    const _Float16* __restrict__ h2t, const float* __restrict__ xx, int* __restrict__ idx) {
  int b = blockIdx.y, i = blockIdx.x, tid = threadIdx.x;
  __shared__ float xi[64];
  __shared__ float dist[NPTS];
  __shared__ float rd[256]; __shared__ int ri[256];
  const _Float16* Hb = h2t + (size_t)b * NPTS * 64;
  for (int c = tid; c < 64; c += 256) xi[c] = (float)Hb[(size_t)i * 64 + c];
  __syncthreads();
  float xxi = xx[b * NPTS + i];
  for (int j = tid; j < NPTS; j += 256) {
    const _Float16* hj = Hb + (size_t)j * 64;
    float dot = 0.f;
#pragma unroll 8
    for (int c = 0; c < 64; ++c) dot += xi[c] * (float)hj[c];
    dist[j] = xxi - 2.f * dot + xx[b * NPTS + j];
  }
  __syncthreads();
  select_min_k(dist, rd, ri, tid, idx + ((size_t)b * NPTS + i) * KNN, KNN, NPTS);
}

// ======================= kNN over raw xyz ===================================
__global__ void __launch_bounds__(256) knn3_kernel(
    const float* __restrict__ x, int* __restrict__ idx) {
  int b = blockIdx.y, i = blockIdx.x, tid = threadIdx.x;
  __shared__ float dist[NPTS];
  __shared__ float rd[256]; __shared__ int ri[256];
  const float* X = x + (size_t)b * 3 * NPTS;
  float x0 = X[i], x1 = X[NPTS + i], x2 = X[2 * NPTS + i];
  for (int j = tid; j < NPTS; j += 256) {
    float dx = X[j] - x0, dy = X[NPTS + j] - x1, dz = X[2 * NPTS + j] - x2;
    dist[j] = dx * dx + dy * dy + dz * dz;
  }
  __syncthreads();
  select_min_k(dist, rd, ri, tid, idx + ((size_t)b * NPTS + i) * KNN, KNN, NPTS);
}

// ======================= fused edgeconv1 + edgeconv2 (wdg1, wdg2) ===========
// 256 thr = 8 waves, wave m owns out-channel tile m for BOTH layers.
// Column tiles are k-major: one WMMA 16x16 tile = one k, 16 points -> max over k
// is an elementwise register max.  Layer-1 tile is passed via 4KB LDS.
__global__ void __launch_bounds__(256) edge12_kernel(
    const _Float16* __restrict__ h2t, const int* __restrict__ idx1,
    const _Float16* __restrict__ wdg1h, const float* __restrict__ bdg1,
    const _Float16* __restrict__ wdg2h, const float* __restrict__ bdg2,
    _Float16* __restrict__ featcat, _Float16* __restrict__ x2t) {
  const int b = blockIdx.y;
  const int n0 = blockIdx.x * 16;
  const int w = threadIdx.x >> 5;        // M-tile 0..7
  const int lane = threadIdx.x & 31;
  const int col = lane & 15;
  const int half = lane >> 4;
  const int n = n0 + col;
  const _Float16* Hb = h2t + (size_t)b * NPTS * 64;
  const int* Ib = idx1 + ((size_t)b * NPTS + n) * KNN;

  v16h A1[4], A2[4];
#pragma unroll
  for (int kt = 0; kt < 4; ++kt) {
    int base = kt * 32 + half * 16;
    A1[kt] = load_wmma_a(wdg1h + (size_t)(w * 16 + col) * 128, base);
    A2[kt] = load_wmma_a(wdg2h + (size_t)(w * 16 + col) * 128, base);
  }
  float bias1[8], bias2[8];
#pragma unroll
  for (int r = 0; r < 8; ++r) {
    int ch = w * 16 + half * 8 + r;
    bias1[r] = bdg1[ch]; bias2[r] = bdg2[ch];
  }

  __shared__ __align__(32) _Float16 g1[16 * 128];
  v8f m1, m2;
#pragma unroll
  for (int r = 0; r < 8; ++r) { m1[r] = -3.0e38f; m2[r] = -3.0e38f; }

  for (int k = 0; k < KNN; ++k) {
    int nbr = Ib[k];
    v8f c1 = {};
#pragma unroll
    for (int kt = 0; kt < 4; ++kt) {
      int c0 = kt * 32 + half * 16;
      v16h Bv;
      if (c0 < 64) {                      // channels 0..63: nbr - ctr
        v16h va = *(const v16h*)(Hb + (size_t)nbr * 64 + c0);
        v16h vb = *(const v16h*)(Hb + (size_t)n * 64 + c0);
        Bv = va - vb;
      } else {                            // channels 64..127: ctr
        Bv = *(const v16h*)(Hb + (size_t)n * 64 + (c0 - 64));
      }
      c1 = __builtin_amdgcn_wmma_f32_16x16x32_f16(false, A1[kt], false, Bv,
                                                  (short)0, c1, false, false);
    }
    v8h o1;
#pragma unroll
    for (int r = 0; r < 8; ++r) {
      float v = lrelu(c1[r] + bias1[r]);
      m1[r] = fmaxf(m1[r], v);
      o1[r] = (_Float16)v;
    }
    __syncthreads();                      // protect g1 vs previous k's reads
    *(v8h*)(g1 + col * 128 + w * 16 + half * 8) = o1;
    __syncthreads();
    v8f c2 = {};
#pragma unroll
    for (int kt = 0; kt < 4; ++kt) {
      v16h Bv = *(const v16h*)(g1 + col * 128 + kt * 32 + half * 16);
      c2 = __builtin_amdgcn_wmma_f32_16x16x32_f16(false, A2[kt], false, Bv,
                                                  (short)0, c2, false, false);
    }
#pragma unroll
    for (int r = 0; r < 8; ++r) m2[r] = fmaxf(m2[r], lrelu(c2[r] + bias2[r]));
  }

  v8h s1, s2;
#pragma unroll
  for (int r = 0; r < 8; ++r) { s1[r] = (_Float16)m1[r]; s2[r] = (_Float16)m2[r]; }
  _Float16* fc = featcat + ((size_t)b * NPTS + n) * 512;
  *(v8h*)(fc + w * 16 + half * 8) = s1;                    // x1 -> ch 0..127
  *(v8h*)(fc + 128 + w * 16 + half * 8) = s2;              // x2 -> ch 128..255
  *(v8h*)(x2t + ((size_t)b * NPTS + n) * 128 + w * 16 + half * 8) = s2;
}

// ======================= edgeconv3 (wsn1, 256->256) =========================
// 512 thr = 16 waves; feat2 column tile for current k staged in 8KB LDS.
__global__ void __launch_bounds__(512) edge3_kernel(
    const _Float16* __restrict__ x2t, const int* __restrict__ idx2,
    const _Float16* __restrict__ wsn1h, const float* __restrict__ bsn1,
    _Float16* __restrict__ featcat) {
  const int b = blockIdx.y;
  const int n0 = blockIdx.x * 16;
  const int w = threadIdx.x >> 5;        // M-tile 0..15
  const int lane = threadIdx.x & 31;
  const int col = lane & 15;
  const int half = lane >> 4;
  const int n = n0 + col;
  const _Float16* Xb = x2t + (size_t)b * NPTS * 128;

  v16h A[8];
#pragma unroll
  for (int kt = 0; kt < 8; ++kt)
    A[kt] = load_wmma_a(wsn1h + (size_t)(w * 16 + col) * 256, kt * 32 + half * 16);
  float bias[8];
#pragma unroll
  for (int r = 0; r < 8; ++r) bias[r] = bsn1[w * 16 + half * 8 + r];

  __shared__ __align__(32) _Float16 f2[16 * 256];
  v8f mx;
#pragma unroll
  for (int r = 0; r < 8; ++r) mx[r] = -3.0e38f;

  // cooperative staging map: thread -> (point sp, 8-channel chunk sc)
  const int sp = threadIdx.x & 15;
  const int sc = (threadIdx.x >> 4) * 8;            // 0..248
  const int sn = n0 + sp;
  const int* Ib2 = idx2 + ((size_t)b * NPTS + sn) * KNN;

  for (int k = 0; k < KNN; ++k) {
    __syncthreads();
    v8h val;
    if (sc < 128) {
      int nbr = Ib2[k];
      v8h va = *(const v8h*)(Xb + (size_t)nbr * 128 + sc);
      v8h vb = *(const v8h*)(Xb + (size_t)sn * 128 + sc);
      val = va - vb;
    } else {
      val = *(const v8h*)(Xb + (size_t)sn * 128 + (sc - 128));
    }
    *(v8h*)(f2 + sp * 256 + sc) = val;
    __syncthreads();
    v8f c = {};
#pragma unroll
    for (int kt = 0; kt < 8; ++kt) {
      v16h Bv = *(const v16h*)(f2 + col * 256 + kt * 32 + half * 16);
      c = __builtin_amdgcn_wmma_f32_16x16x32_f16(false, A[kt], false, Bv,
                                                 (short)0, c, false, false);
    }
#pragma unroll
    for (int r = 0; r < 8; ++r) mx[r] = fmaxf(mx[r], lrelu(c[r] + bias[r]));
  }
  v8h s;
#pragma unroll
  for (int r = 0; r < 8; ++r) s[r] = (_Float16)mx[r];
  *(v8h*)(featcat + ((size_t)b * NPTS + n) * 512 + 256 + w * 16 + half * 8) = s; // x3
}

// ======================= final conv (w3: 512 -> 1024) =======================
__global__ void __launch_bounds__(256) conv3_kernel(
    const _Float16* __restrict__ featcat, const _Float16* __restrict__ w3h,
    const float* __restrict__ b3, float* __restrict__ emb) {
  const int b = blockIdx.z;
  const int n0 = blockIdx.x * 16;
  const int mg = blockIdx.y;             // 0..7
  const int w = threadIdx.x >> 5;
  const int lane = threadIdx.x & 31;
  const int col = lane & 15;
  const int half = lane >> 4;
  const int mt = mg * 8 + w;             // 0..63
  const int n = n0 + col;
  const _Float16* Fb = featcat + ((size_t)b * NPTS + n) * 512;
  const _Float16* Wr = w3h + (size_t)(mt * 16 + col) * 512;

  v8f acc = {};
#pragma unroll
  for (int kt = 0; kt < 16; ++kt) {
    v16h Av = load_wmma_a(Wr, kt * 32 + half * 16);
    v16h Bv = *(const v16h*)(Fb + kt * 32 + half * 16);
    acc = __builtin_amdgcn_wmma_f32_16x16x32_f16(false, Av, false, Bv,
                                                 (short)0, acc, false, false);
  }
#pragma unroll
  for (int r = 0; r < 8; ++r) {
    int o = mt * 16 + half * 8 + r;
    emb[((size_t)b * 1024 + o) * NPTS + n] = lrelu(acc[r] + b3[o]);
  }
}

// ======================= FPS (32 samples) ===================================
__global__ void __launch_bounds__(256) fps_kernel(const float* __restrict__ src,
                                                  int* __restrict__ sidx) {
  int b = blockIdx.x, tid = threadIdx.x;
  __shared__ float dist[NPTS];
  __shared__ float rd[256]; __shared__ int ri[256];
  __shared__ int farS;
  const float* X = src + (size_t)b * 3 * NPTS;
  for (int j = tid; j < NPTS; j += 256) dist[j] = 1.0e10f;
  if (tid == 0) farS = 0;
  __syncthreads();
  for (int it = 0; it < 32; ++it) {
    int far = farS;
    if (tid == 0) sidx[b * 32 + it] = far;
    float cx = X[far], cy = X[NPTS + far], cz = X[2 * NPTS + far];
    for (int j = tid; j < NPTS; j += 256) {
      float dx = X[j] - cx, dy = X[NPTS + j] - cy, dz = X[2 * NPTS + j] - cz;
      dist[j] = fminf(dist[j], dx * dx + dy * dy + dz * dz);
    }
    __syncthreads();
    float bd = -1.f; int bi = 0x7fffffff;
    for (int j = tid; j < NPTS; j += 256) {
      float d = dist[j];
      if (d > bd || (d == bd && j < bi)) { bd = d; bi = j; }
    }
    rd[tid] = bd; ri[tid] = bi;
    __syncthreads();
    for (int s = 128; s > 0; s >>= 1) {
      if (tid < s) {
        float d2 = rd[tid + s]; int i2 = ri[tid + s];
        if (d2 > rd[tid] || (d2 == rd[tid] && i2 < ri[tid])) { rd[tid] = d2; ri[tid] = i2; }
      }
      __syncthreads();
    }
    if (tid == 0) farS = ri[0];
    __syncthreads();
  }
}

// ======================= triplet term (kFN + dp/dn) =========================
__global__ void __launch_bounds__(256) trip_kernel(
    const float* __restrict__ src, const int* __restrict__ sidx,
    const float* __restrict__ semb, const float* __restrict__ temb,
    float* __restrict__ trip_out) {
  __shared__ float px[4][32], py[4][32], pz[4][32];
  __shared__ int far8[4][32][8];
  __shared__ float dmat[32][32];
  __shared__ float tsum[1152];
  int tid = threadIdx.x;
  if (tid < 128) {
    int b = tid >> 5, i = tid & 31;
    int s = sidx[b * 32 + i];
    const float* X = src + (size_t)b * 3 * NPTS;
    px[b][i] = X[s]; py[b][i] = X[NPTS + s]; pz[b][i] = X[2 * NPTS + s];
  }
  __syncthreads();
  for (int b = 0; b < 4; ++b) {
    for (int e = tid; e < 1024; e += 256) {
      int i = e >> 5, j = e & 31;
      float dx = px[b][i] - px[b][j], dy = py[b][i] - py[b][j], dz = pz[b][i] - pz[b][j];
      dmat[i][j] = dx * dx + dy * dy + dz * dz;
    }
    __syncthreads();
    if (tid < 32) {                       // top-8 farthest per row
      int i = tid;
      float dloc[32];
      for (int j = 0; j < 32; ++j) dloc[j] = dmat[i][j];
      for (int t = 0; t < 8; ++t) {
        float bd = -1.f; int bj = 0;
        for (int j = 0; j < 32; ++j)
          if (dloc[j] > bd) { bd = dloc[j]; bj = j; }
        far8[b][i][t] = bj; dloc[bj] = -2.f;
      }
    }
    __syncthreads();
  }
  // task t = (b, i, j): j<8 -> dn pair, j==8 -> dp pair
  for (int t = tid; t < 1152; t += 256) {
    int b = t / 288, rem = t % 288, i = rem / 9, j = rem % 9;
    int si = sidx[b * 32 + i];
    int tj = (j == 8) ? si : sidx[b * 32 + far8[b][i][j]];
    const float* S = semb + (size_t)b * 1024 * NPTS + si;
    const float* T = temb + (size_t)b * 1024 * NPTS + tj;
    float s = 0.f;
    for (int c = 0; c < 1024; ++c) {
      float d = S[(size_t)c * NPTS] - T[(size_t)c * NPTS];
      s += d * d;
    }
    tsum[t] = s;
  }
  __syncthreads();
  if (tid < 128) {
    int b = tid >> 5, i = tid & 31;
    int base = b * 288 + i * 9;
    float dn = 0.f;
    for (int j = 0; j < 8; ++j) dn += tsum[base + j];
    dn /= (1024.f * 8.f);
    float dp = tsum[base + 8] / 1024.f;
    float tr = 1.f - dn / (1.f + dp);
    trip_out[tid] = tr > 0.f ? tr : 0.f;
  }
}

// ======================= (||emb_col|| - 1)^2 per point ======================
__global__ void __launch_bounds__(256) lensq_kernel(const float* __restrict__ emb,
                                                    float* __restrict__ lsq) {
  int tid = threadIdx.x, grp = tid >> 5, lane = tid & 31;
  int p = blockIdx.x * 8 + grp;          // 0..B*N-1
  int b = p >> 12, n = p & (NPTS - 1);
  const float* E = emb + (size_t)b * 1024 * NPTS + n;
  float s = 0.f;
  for (int c = lane; c < 1024; c += 32) {
    float v = E[(size_t)c * NPTS];
    s += v * v;
  }
  __shared__ float red[256];
  red[tid] = s; __syncthreads();
  for (int st = 16; st > 0; st >>= 1) {
    if (lane < st) red[tid] += red[tid + st];
    __syncthreads();
  }
  if (lane == 0) { float l = sqrtf(red[grp * 32]); float d = l - 1.f; lsq[p] = d * d; }
}

// ======================= MSE/MAE partial sums ===============================
__global__ void __launch_bounds__(256) msemae_kernel(const float* __restrict__ a,
                                                     const float* __restrict__ b,
                                                     float* __restrict__ pm,
                                                     float* __restrict__ pa) {
  const size_t total = EMB_ELEMS;
  float sm = 0.f, sa = 0.f;
  for (size_t i = (size_t)blockIdx.x * 256 + threadIdx.x; i < total;
       i += (size_t)gridDim.x * 256) {
    float d = a[i] - b[i];
    sm += d * d; sa += fabsf(d);
  }
  __shared__ float r1[256], r2[256];
  r1[threadIdx.x] = sm; r2[threadIdx.x] = sa; __syncthreads();
  for (int st = 128; st > 0; st >>= 1) {
    if (threadIdx.x < st) { r1[threadIdx.x] += r1[threadIdx.x + st]; r2[threadIdx.x] += r2[threadIdx.x + st]; }
    __syncthreads();
  }
  if (threadIdx.x == 0) { pm[blockIdx.x] = r1[0]; pa[blockIdx.x] = r2[0]; }
}

// ======================= final scalar assembly ==============================
__global__ void __launch_bounds__(256) final_kernel(
    const float* __restrict__ pm, const float* __restrict__ pa, int np,
    const float* __restrict__ slsq, const float* __restrict__ tlsq,
    const float* __restrict__ trip, float* __restrict__ out) {
  __shared__ float red[256];
  int tid = threadIdx.x;
  float sums[5];
  const float* ptrs[5] = { pm, pa, slsq, tlsq, trip };
  int ns[5] = { np, np, BATCH * NPTS, BATCH * NPTS, 128 };
  for (int q = 0; q < 5; ++q) {
    float s = 0.f;
    for (int i = tid; i < ns[q]; i += 256) s += ptrs[q][i];
    red[tid] = s; __syncthreads();
    for (int st = 128; st > 0; st >>= 1) {
      if (tid < st) red[tid] += red[tid + st];
      __syncthreads();
    }
    sums[q] = red[0];
    __syncthreads();
  }
  if (tid == 0) {
    const double tot = (double)EMB_ELEMS;
    float mse = (float)((double)sums[0] / tot * (double)BATCH);
    float mae = (float)((double)sums[1] / tot * (double)BATCH);
    float ln1 = sqrtf(sums[2] / (float)(BATCH * NPTS));
    float ln2 = sqrtf(sums[3] / (float)(BATCH * NPTS));
    float loss = sums[4] / 128.f + (ln1 + ln2) * 0.5f * 0.03f;
    out[0] = loss; out[1] = mse; out[2] = mae;
  }
}

// ============================================================================
extern "C" void kernel_launch(void* const* d_in, const int* in_sizes, int n_in,
                              void* d_out, int out_size, void* d_ws, size_t ws_size,
                              hipStream_t stream) {
  (void)in_sizes; (void)n_in; (void)out_size; (void)ws_size;
  const float* src  = (const float*)d_in[0];
  const float* tgt  = (const float*)d_in[1];
  const float* w1   = (const float*)d_in[2];
  const float* b1   = (const float*)d_in[3];
  const float* w2   = (const float*)d_in[4];
  const float* b2   = (const float*)d_in[5];
  const float* wdg1 = (const float*)d_in[6];
  const float* bdg1 = (const float*)d_in[7];
  const float* wdg2 = (const float*)d_in[8];
  const float* bdg2 = (const float*)d_in[9];
  const float* wsn1 = (const float*)d_in[10];
  const float* bsn1 = (const float*)d_in[11];
  const float* w3   = (const float*)d_in[12];
  const float* b3   = (const float*)d_in[13];
  float* out = (float*)d_out;

  char* ws = (char*)d_ws;
  size_t off = 0;
  auto carve = [&](size_t bytes) -> char* {
    char* p = ws + off;
    off += (bytes + 255) & ~(size_t)255;
    return p;
  };
  _Float16* wdg1h = (_Float16*)carve(128 * 128 * 2);
  _Float16* wdg2h = (_Float16*)carve(128 * 128 * 2);
  _Float16* wsn1h = (_Float16*)carve(256 * 256 * 2);
  _Float16* w3h   = (_Float16*)carve(1024 * 512 * 2);
  _Float16* h2t   = (_Float16*)carve((size_t)BATCH * NPTS * 64 * 2);
  float*    xxb   = (float*)   carve((size_t)BATCH * NPTS * 4);
  int*      idx1  = (int*)     carve((size_t)BATCH * NPTS * KNN * 4);
  int*      idx2  = (int*)     carve((size_t)BATCH * NPTS * KNN * 4);
  _Float16* x2t   = (_Float16*)carve((size_t)BATCH * NPTS * 128 * 2);
  _Float16* fcat  = (_Float16*)carve((size_t)BATCH * NPTS * 512 * 2);
  int*      sidx  = (int*)     carve(BATCH * 32 * 4);
  float*    tripb = (float*)   carve(128 * 4);
  float*    slsq  = (float*)   carve((size_t)BATCH * NPTS * 4);
  float*    tlsq  = (float*)   carve((size_t)BATCH * NPTS * 4);
  float*    pm    = (float*)   carve(1024 * 4);
  float*    pa    = (float*)   carve(1024 * 4);

  // weight conversions
  prep_f16_kernel<<<(128 * 128 + 255) / 256, 256, 0, stream>>>(wdg1, wdg1h, 128 * 128);
  prep_f16_kernel<<<(128 * 128 + 255) / 256, 256, 0, stream>>>(wdg2, wdg2h, 128 * 128);
  prep_f16_kernel<<<(256 * 256 + 255) / 256, 256, 0, stream>>>(wsn1, wsn1h, 256 * 256);
  prep_f16_kernel<<<(1024 * 512 + 255) / 256, 256, 0, stream>>>(w3, w3h, 1024 * 512);

  for (int pass = 0; pass < 2; ++pass) {
    const float* X = pass ? tgt : src;
    float* emb = out + (size_t)pass * EMB_ELEMS;
    conv12_kernel<<<(BATCH * NPTS) / 256, 256, 0, stream>>>(X, w1, b1, w2, b2, h2t, xxb);
    knn64_kernel<<<dim3(NPTS, BATCH), 256, 0, stream>>>(h2t, xxb, idx1);
    knn3_kernel<<<dim3(NPTS, BATCH), 256, 0, stream>>>(X, idx2);
    edge12_kernel<<<dim3(NPTS / 16, BATCH), 256, 0, stream>>>(
        h2t, idx1, wdg1h, bdg1, wdg2h, bdg2, fcat, x2t);
    edge3_kernel<<<dim3(NPTS / 16, BATCH), 512, 0, stream>>>(x2t, idx2, wsn1h, bsn1, fcat);
    conv3_kernel<<<dim3(NPTS / 16, 8, BATCH), 256, 0, stream>>>(fcat, w3h, b3, emb);
  }

  const float* semb = out;
  const float* temb = out + EMB_ELEMS;
  fps_kernel<<<BATCH, 256, 0, stream>>>(src, sidx);
  trip_kernel<<<1, 256, 0, stream>>>(src, sidx, semb, temb, tripb);
  lensq_kernel<<<(BATCH * NPTS) / 8, 256, 0, stream>>>(semb, slsq);
  lensq_kernel<<<(BATCH * NPTS) / 8, 256, 0, stream>>>(temb, tlsq);
  msemae_kernel<<<1024, 256, 0, stream>>>(semb, temb, pm, pa);
  final_kernel<<<1, 256, 0, stream>>>(pm, pa, 1024, slsq, tlsq, tripb,
                                      out + 2 * EMB_ELEMS);
}
